// MixtureOfExperts_5385888989689
// MI455X (gfx1250) — matrix operand order repaired and checked by reading, not compiled, not run on hardware
//
#include <hip/hip_runtime.h>
#include <math.h>

// ---------------------------------------------------------------------------
// MoE top-2 routed FFN for gfx1250 (CDNA5, wave32, WMMA bf16, async-to-LDS)
// ---------------------------------------------------------------------------

#define B_TOK   8192
#define D       1024      // D_IN == D_EXP
#define N_EXP   8
#define BM      64        // token rows per workgroup in FFN kernel
#define HS      1032      // LDS row stride (elements) for h tile; 2*HS % 16 == 0
#define KC      128       // K staging chunk for x tile
#define XS      136       // LDS row stride for x stage tile; 2*XS % 16 == 0

typedef __bf16 bf16_t;
typedef __attribute__((ext_vector_type(16))) __bf16 v16bf;
typedef __attribute__((ext_vector_type(8)))  float  v8f;
typedef __attribute__((ext_vector_type(4)))  __bf16 bf16x4;

union Frag {
    v16bf v;
    uint4 u[2];
};

// CDNA5 async global->LDS copy (tracked by ASYNCcnt).
__device__ __forceinline__ void async_ld_b128(unsigned int lds_addr,
                                              const void* gaddr) {
    asm volatile("global_load_async_to_lds_b128 %0, %1, off"
                 :: "v"(lds_addr),
                    "v"((unsigned long long)(uintptr_t)gaddr)
                 : "memory");
}
__device__ __forceinline__ void wait_async0() {
    asm volatile("s_wait_asynccnt 0" ::: "memory");
}

// Branchless erf (Abramowitz-Stegun 7.1.26, |eps| <= 1.5e-7): uses the
// hardware RCP/EXP transcendental units, no exec divergence (~13 VALU ops
// vs ~30+ divergent ops for libm erff).
__device__ __forceinline__ float fast_erf(float x) {
    float ax = fabsf(x);
    float t  = __builtin_amdgcn_rcpf(fmaf(0.3275911f, ax, 1.0f));
    float p  = fmaf(t, 1.061405429f, -1.453152027f);
    p = fmaf(t, p, 1.421413741f);
    p = fmaf(t, p, -0.284496736f);
    p = fmaf(t, p, 0.254829592f);
    p = p * t;
    float e = __expf(-ax * ax);          // v_exp_f32
    float r = fmaf(-p, e, 1.0f);
    return copysignf(r, x);
}

__device__ __forceinline__ float gelu_exact(float v) {
    return 0.5f * v * (1.0f + fast_erf(v * 0.70710678118654752f));
}

// ---------------------------------------------------------------------------
// K0: zero expert counters
// ---------------------------------------------------------------------------
__global__ void moe_zero_counts(int* counts) {
    if (threadIdx.x < N_EXP) counts[threadIdx.x] = 0;
}

// ---------------------------------------------------------------------------
// K1: transpose+convert w1/w2 (fp32, [e][k][n]) -> bf16 [e][n][k]
// ---------------------------------------------------------------------------
__global__ __launch_bounds__(256)
void moe_wt_cvt(const float* __restrict__ w1, const float* __restrict__ w2,
                bf16_t* __restrict__ w1t, bf16_t* __restrict__ w2t) {
    __shared__ float tile[32][33];
    const int m = blockIdx.y;  // 0..7 -> w1 expert, 8..15 -> w2 expert
    const float*  src = (m < N_EXP) ? (w1 + (size_t)m * D * D)
                                    : (w2 + (size_t)(m - N_EXP) * D * D);
    bf16_t*       dst = (m < N_EXP) ? (w1t + (size_t)m * D * D)
                                    : (w2t + (size_t)(m - N_EXP) * D * D);
    const int tn = blockIdx.x & 31;   // n tile
    const int tk = blockIdx.x >> 5;   // k tile
    const int lx = threadIdx.x;       // 0..31
    const int ly = threadIdx.y;       // 0..7
#pragma unroll
    for (int i = 0; i < 4; ++i) {
        int k = tk * 32 + ly + i * 8;
        int n = tn * 32 + lx;
        tile[ly + i * 8][lx] = src[(size_t)k * D + n];
    }
    __syncthreads();
#pragma unroll
    for (int i = 0; i < 4; ++i) {
        int n = tn * 32 + ly + i * 8;
        int k = tk * 32 + lx;
        dst[(size_t)n * D + k] = (bf16_t)tile[lx][ly + i * 8];
    }
}

// ---------------------------------------------------------------------------
// K2: convert x fp32 -> bf16
// ---------------------------------------------------------------------------
__global__ __launch_bounds__(256)
void moe_x_cvt(const float* __restrict__ x, bf16_t* __restrict__ xb) {
    size_t i = ((size_t)blockIdx.x * blockDim.x + threadIdx.x) * 4;
    float4 v = *(const float4*)(x + i);
    bf16x4 o = { (bf16_t)v.x, (bf16_t)v.y, (bf16_t)v.z, (bf16_t)v.w };
    *(bf16x4*)(xb + i) = o;
}

// ---------------------------------------------------------------------------
// K3: gating. One wave32 per token: exact top-2 softmax over 8 logits.
// ---------------------------------------------------------------------------
__global__ __launch_bounds__(256)
void moe_gate(const float* __restrict__ x, const float* __restrict__ gw,
              float* __restrict__ gate_out, int* __restrict__ top_idx,
              float* __restrict__ top_w, int* __restrict__ counts) {
    const int token = blockIdx.x * 8 + (threadIdx.x >> 5);
    const int lane  = threadIdx.x & 31;
    const float* xr = x + (size_t)token * D;
    float p[N_EXP];
#pragma unroll
    for (int e = 0; e < N_EXP; ++e) p[e] = 0.0f;
    for (int d = lane; d < D; d += 32) {
        float xv = xr[d];
        const float* g = gw + (size_t)d * N_EXP;
#pragma unroll
        for (int e = 0; e < N_EXP; ++e) p[e] += xv * g[e];
    }
#pragma unroll
    for (int off = 16; off >= 1; off >>= 1) {
#pragma unroll
        for (int e = 0; e < N_EXP; ++e) p[e] += __shfl_xor(p[e], off, 32);
    }
    if (lane == 0) {
        int i0 = 0; float v0 = p[0];
#pragma unroll
        for (int e = 1; e < N_EXP; ++e) { if (p[e] > v0) { v0 = p[e]; i0 = e; } }
        int i1 = -1; float v1 = -3.4e38f;
#pragma unroll
        for (int e = 0; e < N_EXP; ++e) {
            if (e != i0 && p[e] > v1) { v1 = p[e]; i1 = e; }
        }
        float ex = __expf(v1 - v0);       // v0 >= v1 -> ex in (0,1]
        float den = 1.0f + ex;
        float w0 = 1.0f / den;
        float w1 = ex / den;
        float* go = gate_out + (size_t)token * N_EXP;
        float4 z = {0.f, 0.f, 0.f, 0.f};
        *(float4*)go = z;
        *(float4*)(go + 4) = z;
        go[i0] = w0;
        go[i1] = w1;
        top_idx[token * 2 + 0] = i0;
        top_idx[token * 2 + 1] = i1;
        top_w[token * 2 + 0] = w0;
        top_w[token * 2 + 1] = w1;
        atomicAdd(&counts[i0], 1);
        atomicAdd(&counts[i1], 1);
    }
}

// ---------------------------------------------------------------------------
// K4: exclusive scan of 8 counts
// ---------------------------------------------------------------------------
__global__ void moe_scan(const int* __restrict__ counts,
                         int* __restrict__ basep, int* __restrict__ cursor) {
    if (threadIdx.x == 0) {
        int acc = 0;
        for (int e = 0; e < N_EXP; ++e) {
            basep[e] = acc;
            cursor[e] = acc;
            acc += counts[e];
        }
    }
}

// ---------------------------------------------------------------------------
// K5: scatter token-expert pairs into per-expert segments
// ---------------------------------------------------------------------------
__global__ __launch_bounds__(256)
void moe_scatter(const int* __restrict__ top_idx, int* __restrict__ cursor,
                 int* __restrict__ routed) {
    int b = blockIdx.x * blockDim.x + threadIdx.x;
#pragma unroll
    for (int k = 0; k < 2; ++k) {
        int e = top_idx[b * 2 + k];
        int pos = atomicAdd(&cursor[e], 1);
        routed[pos] = (b << 1) | k;  // token in high bits, slot k in bit 0
    }
}

// ---------------------------------------------------------------------------
// K6: fused routed FFN.  One workgroup = (expert, 64 routed rows).
//     Each wave owns a disjoint 64-column N-strip and register-blocks all
//     four 16-row M-tiles (acc[4][4] f32).  Weight columns are loaded by
//     exactly one wave (no cross-wave redundancy); A tiles are shared via
//     LDS: x staged in KC-chunks with global_load_async_to_lds_b128, and
//     the GELU'd h tile lives in LDS between the two GEMMs.
// ---------------------------------------------------------------------------
__global__ __launch_bounds__(256)
void moe_ffn(const bf16_t* __restrict__ xb,    // [B_TOK][D] bf16
             const bf16_t* __restrict__ w1t,   // [E][n][k] bf16
             const bf16_t* __restrict__ w2t,   // [E][n][k] bf16
             const float*  __restrict__ b1,    // [E][D]
             const float*  __restrict__ b2,    // [E][D]
             const int*    __restrict__ routed,// [2*B_TOK]
             const int*    __restrict__ basep, // [E]
             const int*    __restrict__ counts,// [E]
             const float*  __restrict__ top_w, // [B_TOK][2]
             float*        __restrict__ slot)  // [2][B_TOK][D]
{
    __shared__ bf16_t h_s[BM * HS];   // 132,096 B
    __shared__ bf16_t x_s[BM * XS];   //  17,408 B
    __shared__ int routed_s[BM];

    const int e    = blockIdx.y;
    const int cnt  = counts[e];
    const int mblk = blockIdx.x;
    if (mblk * BM >= cnt) return;

    const int rbase = basep[e] + mblk * BM;
    const int valid = cnt - mblk * BM;
    const int tid = threadIdx.x;
    if (tid < BM) {
        routed_s[tid] = (tid < valid) ? routed[rbase + tid] : -1;
    }
    __syncthreads();

    const int wave = tid >> 5;       // 0..7: owns N columns [wave*64, +64)
    const int lane = tid & 31;
    const int lr   = lane & 15;
    const int lhi  = lane >> 4;      // 0/1

    // staging assignment: thread -> (row, 32-element part)
    const int srow  = tid & 63;
    const int spart = tid >> 6;      // 0..3
    const int srv   = routed_s[srow];
    const bf16_t* sgrow = xb + (size_t)((srv >= 0) ? (srv >> 1) : 0) * D
                             + spart * 32;
    const unsigned int slds =
        (unsigned int)(uintptr_t)(x_s + srow * XS + spart * 32);

    const bf16_t* w1e = w1t + (size_t)e * D * D;
    const float*  b1e = b1 + (size_t)e * D;

    // =========================== GEMM1 ====================================
    for (int pass = 0; pass < 2; ++pass) {
        const int ncol = pass * 512 + wave * 64;
        v8f acc[4][4] = {};
        for (int kk = 0; kk < D; kk += KC) {
            __syncthreads();   // previous chunk fully consumed
            // async-stage x[rows][kk..kk+KC) -> x_s
#pragma unroll
            for (int j = 0; j < 4; ++j) {
                async_ld_b128(slds + j * 16, sgrow + kk + j * 8);
            }
            wait_async0();
            __syncthreads();   // staged chunk visible to all waves

#pragma unroll
            for (int ks = 0; ks < KC; ks += 32) {
                // B fragments: lane holds column n, 16 consecutive K
                const int kb = kk + ks + (lhi ? 16 : 0);
                Frag bf[4];
#pragma unroll
                for (int t = 0; t < 4; ++t) {
                    const int n = ncol + t * 16 + lr;
                    const bf16_t* bp = w1e + (size_t)n * D + kb;
                    bf[t].u[0] = *(const uint4*)(bp);
                    bf[t].u[1] = *(const uint4*)(bp + 8);
                }
                const int ka = ks + (lhi ? 16 : 0);
#pragma unroll
                for (int m = 0; m < 4; ++m) {
                    const bf16_t* ap = x_s + (m * 16 + lr) * XS + ka;
                    Frag af;
                    af.u[0] = *(const uint4*)(ap);        // ds_load_b128
                    af.u[1] = *(const uint4*)(ap + 8);
#pragma unroll
                    for (int t = 0; t < 4; ++t) {
                        acc[m][t] = __builtin_amdgcn_wmma_f32_16x16x32_bf16(
                            false, af.v, false, bf[t].v, (short)0,
                            acc[m][t], false, false);
                    }
                }
            }
        }
        // epilogue: bias + branchless-erf GELU -> bf16 tile in LDS
#pragma unroll
        for (int t = 0; t < 4; ++t) {
            const int n = ncol + t * 16 + lr;
            const float bias = b1e[n];
#pragma unroll
            for (int m = 0; m < 4; ++m) {
#pragma unroll
                for (int r = 0; r < 8; ++r) {
                    const int row = m * 16 + r + (lhi ? 8 : 0);
                    float v = acc[m][t][r] + bias;
                    h_s[row * HS + n] = (bf16_t)gelu_exact(v);
                }
            }
        }
    }
    __syncthreads();   // h tile complete

    // =========================== GEMM2 ====================================
    const bf16_t* w2e = w2t + (size_t)e * D * D;
    const float*  b2e = b2 + (size_t)e * D;

    for (int pass = 0; pass < 2; ++pass) {
        const int ncol = pass * 512 + wave * 64;
        v8f acc[4][4] = {};
        for (int kk = 0; kk < D; kk += 32) {
            const int kb = kk + (lhi ? 16 : 0);
            Frag bf[4];
#pragma unroll
            for (int t = 0; t < 4; ++t) {
                const int n = ncol + t * 16 + lr;
                const bf16_t* bp = w2e + (size_t)n * D + kb;
                bf[t].u[0] = *(const uint4*)(bp);
                bf[t].u[1] = *(const uint4*)(bp + 8);
            }
            const int ka = kk + (lhi ? 16 : 0);
#pragma unroll
            for (int m = 0; m < 4; ++m) {
                const bf16_t* ap = h_s + (m * 16 + lr) * HS + ka;
                Frag af;
                af.u[0] = *(const uint4*)(ap);            // ds_load_b128
                af.u[1] = *(const uint4*)(ap + 8);
#pragma unroll
                for (int t = 0; t < 4; ++t) {
                    acc[m][t] = __builtin_amdgcn_wmma_f32_16x16x32_bf16(
                        false, af.v, false, bf[t].v, (short)0,
                        acc[m][t], false, false);
                }
            }
        }
        // epilogue: bias + gate-scale, store to slot buffer
#pragma unroll
        for (int t = 0; t < 4; ++t) {
            const int n = ncol + t * 16 + lr;
            const float bias = b2e[n];
#pragma unroll
            for (int m = 0; m < 4; ++m) {
#pragma unroll
                for (int r = 0; r < 8; ++r) {
                    const int row = m * 16 + r + (lhi ? 8 : 0);
                    const int rv = routed_s[row];
                    if (rv >= 0) {
                        const int token = rv >> 1;
                        const int k = rv & 1;
                        const float g = top_w[token * 2 + k];
                        slot[((size_t)k * B_TOK + token) * D + n] =
                            g * (acc[m][t][r] + bias);
                    }
                }
            }
        }
    }
}

// ---------------------------------------------------------------------------
// K7: combine the two routed slots -> final output
// ---------------------------------------------------------------------------
__global__ __launch_bounds__(256)
void moe_combine(const float* __restrict__ slot, float* __restrict__ out) {
    size_t i = ((size_t)blockIdx.x * blockDim.x + threadIdx.x) * 4;
    float4 a = *(const float4*)(slot + i);
    float4 b = *(const float4*)(slot + (size_t)B_TOK * D + i);
    float4 o = {a.x + b.x, a.y + b.y, a.z + b.z, a.w + b.w};
    *(float4*)(out + i) = o;
}

// ---------------------------------------------------------------------------
// Host launch
// ---------------------------------------------------------------------------
extern "C" void kernel_launch(void* const* d_in, const int* in_sizes, int n_in,
                              void* d_out, int out_size, void* d_ws, size_t ws_size,
                              hipStream_t stream) {
    (void)in_sizes; (void)n_in; (void)out_size; (void)ws_size;

    const float* x  = (const float*)d_in[0];   // (B, D)
    const float* gw = (const float*)d_in[1];   // (D, E)
    const float* w1 = (const float*)d_in[2];   // (E, D, D)
    const float* b1 = (const float*)d_in[3];   // (E, D)
    const float* w2 = (const float*)d_in[4];   // (E, D, D)
    const float* b2 = (const float*)d_in[5];   // (E, D)

    float* out      = (float*)d_out;                        // (B, D)
    float* gate_out = out + (size_t)B_TOK * D;              // (B, E)

    char* ws = (char*)d_ws;
    bf16_t* w1t   = (bf16_t*)(ws);                            // 16 MB
    bf16_t* w2t   = (bf16_t*)(ws + ((size_t)16 << 20));       // 16 MB
    bf16_t* xb    = (bf16_t*)(ws + ((size_t)32 << 20));       // 16 MB
    float*  slot  = (float*) (ws + ((size_t)48 << 20));       // 64 MB
    int*    tidx  = (int*)   (ws + ((size_t)112 << 20));      // 64 KB
    float*  tw    = (float*) (ws + ((size_t)112 << 20) + 65536);
    int*    cnts  = (int*)   (ws + ((size_t)112 << 20) + 131072);
    int*    basep = cnts + 8;
    int*    curs  = cnts + 16;
    int*    routed = cnts + 32;                               // 16384 ints

    moe_zero_counts<<<1, 32, 0, stream>>>(cnts);
    moe_wt_cvt<<<dim3(1024, 16), dim3(32, 8), 0, stream>>>(w1, w2, w1t, w2t);
    moe_x_cvt<<<(B_TOK * D) / (256 * 4), 256, 0, stream>>>(x, xb);
    moe_gate<<<B_TOK / 8, 256, 0, stream>>>(x, gw, gate_out, tidx, tw, cnts);
    moe_scan<<<1, 1, 0, stream>>>(cnts, basep, curs);
    moe_scatter<<<B_TOK / 256, 256, 0, stream>>>(tidx, curs, routed);
    moe_ffn<<<dim3((2 * B_TOK) / BM, N_EXP), 256, 0, stream>>>(
        xb, w1t, w2t, b1, b2, routed, basep, cnts, tw, slot);
    moe_combine<<<(B_TOK * D) / (256 * 4), 256, 0, stream>>>(slot, out);
}
